// SelfAttention_45767171506798
// MI455X (gfx1250) — matrix operand order repaired and checked by reading, not compiled, not run on hardware
//
#include <hip/hip_runtime.h>
#include <hip/hip_bf16.h>
#include <math.h>

typedef __attribute__((ext_vector_type(16))) __bf16 v16bf;
typedef __attribute__((ext_vector_type(8)))  __bf16 v8bf;
typedef __attribute__((ext_vector_type(8)))  float  v8f;
typedef __attribute__((ext_vector_type(4)))  int    v4i;
typedef __attribute__((address_space(1))) v4i as1_v4i;   // global
typedef __attribute__((address_space(3))) v4i as3_v4i;   // LDS

#define SEQ 8192
#define DM  768
#define J3  2304   // 3*DM
#define NKT 24     // DM/32 k-steps

#ifndef __has_builtin
#define __has_builtin(x) 0
#endif
#if __has_builtin(__builtin_amdgcn_global_load_async_to_lds_b128)
#define HAVE_ASYNC_LDS 1
#else
#define HAVE_ASYNC_LDS 0
#endif

// 16-byte global -> LDS copy (async CDNA5 path when available)
__device__ __forceinline__ void cp16(__bf16* dst, const __bf16* src) {
#if HAVE_ASYNC_LDS
    __builtin_amdgcn_global_load_async_to_lds_b128(
        (as1_v4i*)src, (as3_v4i*)dst, 0, 0);
#else
    *(uint4*)dst = *(const uint4*)src;
#endif
}
__device__ __forceinline__ void cp_wait() {
#if HAVE_ASYNC_LDS
# if __has_builtin(__builtin_amdgcn_s_wait_asynccnt)
    __builtin_amdgcn_s_wait_asynccnt(0);
# else
    asm volatile("s_wait_asynccnt 0" ::: "memory");
# endif
#endif
}

// ---------------------------------------------------------------------------
// Kernel A: convert X (f32, row-major) -> Xs (bf16, pre-swizzled WMMA A-frag
// layout): frag (R, ktI) holds rows [16R,16R+16) x cols [32ktI, +32),
// stored as [frag][lane][16 elems] so a lane's A-fragment is one 32B load.
// ---------------------------------------------------------------------------
__global__ __launch_bounds__(128) void conv_x(const float* __restrict__ X,
                                              __bf16* __restrict__ Xs) {
    const int lane = threadIdx.x & 31;
    const int wave = threadIdx.x >> 5;
    const int frag = blockIdx.x * 4 + wave;        // (SEQ/16)*NKT frags
    const int R = frag / NKT, ktI = frag % NKT;
    const int l15 = lane & 15, half = lane >> 4;

    const float* src = X + (size_t)(R * 16 + l15) * DM + ktI * 32;
    const float4* p0 = (const float4*)(src + half * 16);
    const float4* p1 = (const float4*)(src + half * 16 + 8);
    float4 a0 = p0[0], a1 = p0[1], a2 = p1[0], a3 = p1[1];
    v16bf A;
    A[0]=(__bf16)a0.x; A[1]=(__bf16)a0.y; A[2]=(__bf16)a0.z;  A[3]=(__bf16)a0.w;
    A[4]=(__bf16)a1.x; A[5]=(__bf16)a1.y; A[6]=(__bf16)a1.z;  A[7]=(__bf16)a1.w;
    A[8]=(__bf16)a2.x; A[9]=(__bf16)a2.y; A[10]=(__bf16)a2.z; A[11]=(__bf16)a2.w;
    A[12]=(__bf16)a3.x;A[13]=(__bf16)a3.y;A[14]=(__bf16)a3.z; A[15]=(__bf16)a3.w;
    *(v16bf*)(Xs + (size_t)frag * 512 + lane * 16) = A;
}

// Kernel B: convert W (f32) -> Wb (bf16, row-major; B-fragments are
// naturally contiguous 32B runs within a W row).
__global__ __launch_bounds__(256) void conv_w(const float* __restrict__ W,
                                              __bf16* __restrict__ Wb) {
    size_t i = ((size_t)blockIdx.x * 256 + threadIdx.x) * 8;
    float4 a = *(const float4*)(W + i);
    float4 b = *(const float4*)(W + i + 4);
    v8bf o;
    o[0]=(__bf16)a.x; o[1]=(__bf16)a.y; o[2]=(__bf16)a.z; o[3]=(__bf16)a.w;
    o[4]=(__bf16)b.x; o[5]=(__bf16)b.y; o[6]=(__bf16)b.z; o[7]=(__bf16)b.w;
    *(v8bf*)(Wb + i) = o;
}

// ---------------------------------------------------------------------------
// Kernel C: QKV GEMM. out[n,j] = sum_k X[n,k] W[j,k].
// One 16x64 strip per wave: 1 A-frag load + 4 B-frag loads + 4 WMMA per kt.
// Q written pre-scaled into swizzled A-frag layout Qs; K row-major; V transposed.
// ---------------------------------------------------------------------------
__global__ __launch_bounds__(128) void qkv_gemm(const __bf16* __restrict__ Xs,
                                                const __bf16* __restrict__ Wb,
                                                __bf16* __restrict__ Qs,
                                                __bf16* __restrict__ Kb,
                                                __bf16* __restrict__ Vt) {
    const int lane = threadIdx.x & 31;
    const int wave = threadIdx.x >> 5;
    const int strip = blockIdx.x * 4 + wave;       // (SEQ/16)*(J3/64) strips
    const int tM = strip / (J3 / 64);
    const int jS = strip % (J3 / 64);
    const int rowBase = tM * 16;
    const int jBase = jS * 64;
    const int l15 = lane & 15, half = lane >> 4;

    v8f acc[4] = {};
    const __bf16* xfrag = Xs + (size_t)tM * (NKT * 512) + lane * 16;

    #pragma unroll 4
    for (int ktI = 0; ktI < NKT; ++ktI) {
        v16bf A = *(const v16bf*)(xfrag + ktI * 512);
        #pragma unroll
        for (int t = 0; t < 4; ++t) {
            v16bf B = *(const v16bf*)(Wb + (size_t)(jBase + 16 * t + l15) * DM
                                         + ktI * 32 + half * 16);
            acc[t] = __builtin_amdgcn_wmma_f32_16x16x32_bf16(false, A, false, B,
                                                             (short)0, acc[t], false, false);
        }
    }

    const float qscale = 0.03608439182435161f; // 1/sqrt(768)
    #pragma unroll
    for (int t = 0; t < 4; ++t) {
        const int j = jBase + 16 * t + l15;    // branch below is tile-uniform
        #pragma unroll
        for (int r = 0; r < 8; ++r) {
            const int m = rowBase + r + half * 8;
            const float v = acc[t][r];
            if (j < DM) {
                // scatter into swizzled A-frag layout for flash Q
                const int c = j & 31, ktI = j >> 5;
                const int hp = (c >> 4) & 1;
                const int e  = c & 15;
                const int lp = hp * 16 + (m & 15);
                Qs[((size_t)tM * NKT + ktI) * 512 + lp * 16 + e] = (__bf16)(v * qscale);
            } else if (j < 2 * DM) {
                Kb[(size_t)m * DM + (j - DM)] = (__bf16)v;
            } else {
                Vt[(size_t)(j - 2 * DM) * SEQ + m] = (__bf16)v;
            }
        }
    }
}

// ---------------------------------------------------------------------------
// Kernel D: causal flash attention. Block = 4 waves = 16 query rows.
// Per 64-key tile: waves split keys for S (1 frag each, no redundancy),
// exchange row stats via LDS, stage P swizzled in LDS, then split d=768
// (192 cols/wave) for O += P V.
// ---------------------------------------------------------------------------
__global__ __launch_bounds__(128) void flash_attn(const __bf16* __restrict__ Qs,
                                                  const __bf16* __restrict__ Kb,
                                                  const __bf16* __restrict__ Vt,
                                                  float* __restrict__ Out) {
    __shared__ __bf16 Qsm[16 * DM];     // swizzled Q A-frags, 24KB
    __shared__ __bf16 Ksm[64 * DM];     // row-major K tile, 96KB
    __shared__ __bf16 Psm[16 * 64];     // swizzled P A-frags, 2KB
    __shared__ float  MxSm[4 * 16];
    __shared__ float  SxSm[4 * 16];

    const int tid  = threadIdx.x;
    const int lane = tid & 31;
    const int wave = tid >> 5;
    const int l15  = lane & 15;
    const int half = lane >> 4;
    const int qbase = blockIdx.x * 16;
    const int dbase = wave * 192;

    // ---- stage swizzled Q tile: contiguous 24KB region ----
    {
        const __bf16* src = Qs + (size_t)qbase * DM;
        #pragma unroll
        for (int i = 0; i < 12; ++i) {
            const int idx = i * 128 + tid;
            cp16(&Qsm[idx * 8], src + idx * 8);
        }
        cp_wait();
        __syncthreads();
    }

    float mrow[8], lrow[8];
    #pragma unroll
    for (int r = 0; r < 8; ++r) { mrow[r] = -INFINITY; lrow[r] = 0.0f; }
    v8f O[12] = {};

    const int nKv = (qbase + 15) / 64 + 1;
    for (int kb = 0; kb < nKv; ++kb) {
        const int kvbase = kb * 64;

        // ---- stage K tile (64 x 768, contiguous 96KB) ----
        {
            const __bf16* ksrc = Kb + (size_t)kvbase * DM;
            #pragma unroll
            for (int i = 0; i < 48; ++i) {
                const int idx = i * 128 + tid;
                cp16(&Ksm[idx * 8], ksrc + idx * 8);
            }
            cp_wait();
            __syncthreads();
        }

        // ---- S = Q K^T for this wave's 16 keys (single C fragment) ----
        v8f S = {};
        #pragma unroll 4
        for (int ktI = 0; ktI < NKT; ++ktI) {
            v16bf A = *(const v16bf*)(Qsm + (ktI * 32 + lane) * 16);
            v16bf B = *(const v16bf*)(Ksm + (wave * 16 + l15) * DM + ktI * 32 + half * 16);
            S = __builtin_amdgcn_wmma_f32_16x16x32_bf16(false, A, false, B,
                                                        (short)0, S, false, false);
        }

        // ---- causal mask ----
        const int kvw = kvbase + wave * 16;
        if (kvw + 15 > qbase) {
            #pragma unroll
            for (int r = 0; r < 8; ++r)
                if (kvw + l15 > qbase + r + half * 8)
                    S[r] = -INFINITY;
        }

        // ---- per-wave row max -> LDS exchange ----
        float pm[8];
        #pragma unroll
        for (int r = 0; r < 8; ++r) pm[r] = S[r];
        #pragma unroll
        for (int off = 1; off < 16; off <<= 1)
            #pragma unroll
            for (int r = 0; r < 8; ++r)
                pm[r] = fmaxf(pm[r], __shfl_xor(pm[r], off, 32));
        if (l15 == 0)
            #pragma unroll
            for (int r = 0; r < 8; ++r)
                MxSm[wave * 16 + half * 8 + r] = pm[r];
        __syncthreads();

        float mnew[8];
        #pragma unroll
        for (int r = 0; r < 8; ++r) {
            float m = mrow[r];
            #pragma unroll
            for (int w2 = 0; w2 < 4; ++w2)
                m = fmaxf(m, MxSm[w2 * 16 + half * 8 + r]);
            mnew[r] = m;
        }

        // ---- p = exp(S - mnew); partial row sums -> LDS; stage P swizzled ----
        float ps[8];
        #pragma unroll
        for (int r = 0; r < 8; ++r) {
            float p = __expf(S[r] - mnew[r]);
            S[r] = p;
            ps[r] = p;
        }
        #pragma unroll
        for (int off = 1; off < 16; off <<= 1)
            #pragma unroll
            for (int r = 0; r < 8; ++r)
                ps[r] += __shfl_xor(ps[r], off, 32);
        if (l15 == 0)
            #pragma unroll
            for (int r = 0; r < 8; ++r)
                SxSm[wave * 16 + half * 8 + r] = ps[r];
        {
            const int n = wave * 16 + l15;          // key column within 64
            const int kbb = n >> 5, c = n & 31;
            const int f = c;
            #pragma unroll
            for (int r = 0; r < 8; ++r)
                Psm[kbb * 512 + (r + half * 8) * 32 + f] = (__bf16)S[r];
        }
        __syncthreads();

        // ---- combine stats, rescale O ----
        float fac[8];
        #pragma unroll
        for (int r = 0; r < 8; ++r) {
            float rs = SxSm[0 * 16 + half * 8 + r] + SxSm[1 * 16 + half * 8 + r]
                     + SxSm[2 * 16 + half * 8 + r] + SxSm[3 * 16 + half * 8 + r];
            fac[r]  = __expf(mrow[r] - mnew[r]);
            lrow[r] = lrow[r] * fac[r] + rs;
            mrow[r] = mnew[r];
        }
        #pragma unroll
        for (int t = 0; t < 12; ++t)
            #pragma unroll
            for (int r = 0; r < 8; ++r)
                O[t][r] *= fac[r];

        // ---- O += P V (A-frags: single 32B LDS loads; B-frags from Vt) ----
        #pragma unroll
        for (int kbb = 0; kbb < 2; ++kbb) {
            v16bf Ap = *(const v16bf*)(Psm + kbb * 512 + l15 * 32 + half * 16);
            #pragma unroll
            for (int t = 0; t < 12; ++t) {
                v16bf Bv = *(const v16bf*)(Vt + (size_t)(dbase + 16 * t + l15) * SEQ
                                              + kvbase + kbb * 32 + half * 16);
                O[t] = __builtin_amdgcn_wmma_f32_16x16x32_bf16(false, Ap, false, Bv,
                                                               (short)0, O[t], false, false);
            }
        }
        __syncthreads();   // protect Ksm/Psm/MxSm/SxSm for next tile
    }

    // ---- epilogue: O *= 1/l, store f32 ----
    float inv[8];
    #pragma unroll
    for (int r = 0; r < 8; ++r) inv[r] = 1.0f / lrow[r];
    #pragma unroll
    for (int t = 0; t < 12; ++t)
        #pragma unroll
        for (int r = 0; r < 8; ++r) {
            const int m = qbase + r + half * 8;
            const int d = dbase + 16 * t + l15;
            Out[(size_t)m * DM + d] = O[t][r] * inv[r];
        }
}

// ---------------------------------------------------------------------------
extern "C" void kernel_launch(void* const* d_in, const int* in_sizes, int n_in,
                              void* d_out, int out_size, void* d_ws, size_t ws_size,
                              hipStream_t stream) {
    (void)in_sizes; (void)n_in; (void)out_size; (void)ws_size;
    const float* X = (const float*)d_in[0];   // [SEQ, DM] f32
    const float* W = (const float*)d_in[1];   // [3*DM, DM] f32
    float* Out = (float*)d_out;               // [SEQ, DM] f32

    char* ws = (char*)d_ws;
    const size_t xsB = (size_t)SEQ * DM * 2;  // 12.6 MB
    const size_t wbB = (size_t)J3 * DM * 2;   // 3.5 MB
    __bf16* Xs = (__bf16*)(ws);
    __bf16* Wb = (__bf16*)(ws + xsB);
    __bf16* Qs = (__bf16*)(ws + xsB + wbB);
    __bf16* Kb = (__bf16*)(ws + 2 * xsB + wbB);
    __bf16* Vt = (__bf16*)(ws + 3 * xsB + wbB);

    conv_x<<<(SEQ / 16) * NKT / 4, 128, 0, stream>>>(X, Xs);
    conv_w<<<(int)((size_t)J3 * DM / 8 / 256), 256, 0, stream>>>(W, Wb);
    qkv_gemm<<<(SEQ / 16) * (J3 / 64) / 4, 128, 0, stream>>>(Xs, Wb, Qs, Kb, Vt);
    flash_attn<<<SEQ / 16, 128, 0, stream>>>(Qs, Kb, Vt, Out);
}
